// Bigram_73864847556968
// MI455X (gfx1250) — matrix operand order repaired and checked by reading, not compiled, not run on hardware
//
#include <hip/hip_runtime.h>
#include <hip/hip_bf16.h>

#define VOCAB 50257
#define DM    1024
#define NTOK  8192            // B*S = 4*2048
#define KC    (DM / 32)       // 32 K-chunks of 32
#define NPAD  50432           // 197 * 256 (vocab padded to block-tile N)
#define NBLK_N 197            // N blocks of 256
#define NTILES_M (NTOK / 16)  // 512
#define NTILES_N (NPAD / 16)  // 3152

// stage buffer: per K-chunk, 4 A tiles + 16 B tiles, 512 bf16 (1 KB) each
#define STAGE_ELEMS 10240     // 20 tiles * 512 bf16 = 20 KB
#define STAGE_BYTES (STAGE_ELEMS * 2)

typedef __attribute__((ext_vector_type(16))) __bf16          v16bf;
typedef __attribute__((ext_vector_type(16))) unsigned short  v16u;
typedef __attribute__((ext_vector_type(8)))  float           v8f;

__device__ __forceinline__ unsigned short f2bf(float f) {
  unsigned int u = __float_as_uint(f);
  u += 0x7FFFu + ((u >> 16) & 1u);      // round-to-nearest-even
  return (unsigned short)(u >> 16);
}

// Safe streaming log-sum-exp merge: (m,s) <- merge((m,s), (m2,s2)).
__device__ __forceinline__ void lse_merge(float& m, float& s, float m2, float s2) {
  float M  = fmaxf(m, m2);
  float e1 = (m  == -INFINITY) ? 0.0f : __expf(m  - M);
  float e2 = (m2 == -INFINITY) ? 0.0f : __expf(m2 - M);
  s = s * e1 + s2 * e2;
  m = M;
}

// K offsets inside a 32-wide chunk (ISA 7.12.2 16-bit fragment layouts)
__device__ __forceinline__ int kOffA(int v, int h) { return ((v < 4) ? 2 * v : 8 + 2 * v) + 8 * h; }
__device__ __forceinline__ int kOffB(int v, int h) { return ((v < 4) ? 2 * v : 8 + 2 * v) + 8 * h; }

// Async 16B global -> LDS copy (per-lane). ldsByte = wave-relative LDS byte
// address (low 32 bits of a generic __shared__ pointer). Tracked by ASYNCcnt.
__device__ __forceinline__ void async_copy_b128(unsigned ldsByte, const void* gsrc) {
  asm volatile("global_load_async_to_lds_b128 %0, %1, off"
               :: "v"(ldsByte), "v"(gsrc)
               : "memory");
}
__device__ __forceinline__ void wait_asynccnt0() {
  asm volatile("s_wait_asynccnt 0" ::: "memory");
}

// ---------------------------------------------------------------------------
// Gather emb[idx] -> bf16, pre-swizzled into WMMA A-fragment layout.
// ---------------------------------------------------------------------------
__global__ __launch_bounds__(256)
void conv_x_bf16(const int* __restrict__ idx, const float* __restrict__ emb,
                 __bf16* __restrict__ wsA) {
  const int t    = blockIdx.x * 256 + threadIdx.x;
  const int lane = t & 31;
  const int kc   = (t >> 5) & (KC - 1);
  const int tm   = t >> 10;
  if (tm >= NTILES_M) return;
  const int h = lane >> 4, l = lane & 15;
  const float* src = emb + (size_t)idx[tm * 16 + l] * DM + kc * 32;
  v16u o;
#pragma unroll
  for (int v = 0; v < 8; ++v) {
    float2 f = *(const float2*)(src + kOffA(v, h));
    o[2 * v] = f2bf(f.x); o[2 * v + 1] = f2bf(f.y);
  }
  *(v16u*)(wsA + ((size_t)tm * KC + kc) * 512 + lane * 16) = o;
}

// ---------------------------------------------------------------------------
// W -> bf16, pre-swizzled into WMMA B-fragment layout (columns of W^T).
// ---------------------------------------------------------------------------
__global__ __launch_bounds__(256)
void conv_w_bf16(const float* __restrict__ W, __bf16* __restrict__ wsW) {
  const int t    = blockIdx.x * 256 + threadIdx.x;
  const int lane = t & 31;
  const int kc   = (t >> 5) & (KC - 1);
  const int tn   = t >> 10;
  if (tn >= NTILES_N) return;
  const int h = lane >> 4, l = lane & 15;
  int n = tn * 16 + l;
  if (n >= VOCAB) n = VOCAB - 1;                 // padded tiles: duplicate last row
  const float* src = W + (size_t)n * DM + kc * 32;
  v16u o;
#pragma unroll
  for (int v = 0; v < 8; ++v) {
    float2 f = *(const float2*)(src + kOffB(v, h));
    o[2 * v] = f2bf(f.x); o[2 * v + 1] = f2bf(f.y);
  }
  *(v16u*)(wsW + ((size_t)tn * KC + kc) * 512 + lane * 16) = o;
}

// ---------------------------------------------------------------------------
// logits = x @ W^T + b, fused per-(row, N-block) softmax partials.
// Block: 8 waves, tile 64(M) x 256(N); wave tile 32x64 (8 WMMA / K-chunk).
// K-chunks cooperatively staged to LDS via GLOBAL_LOAD_ASYNC_TO_LDS_B128
// (double-buffered, manually 2x-unrolled with literal buffer parity).
// ---------------------------------------------------------------------------
__global__ __launch_bounds__(256)
void bigram_logits_wmma(const __bf16* __restrict__ wsA, const __bf16* __restrict__ wsW,
                        const float* __restrict__ bias, float* __restrict__ logits,
                        float* __restrict__ partials) {
  __shared__ union SharedMem {
    __bf16 stage[2][STAGE_ELEMS];                        // 2 x 20 KB
    struct { float vals[64][260]; float pm[64][4]; float ps[64][4]; } ep;
  } sh;

  const int tid  = threadIdx.x;
  const int lane = tid & 31;
  const int wave = tid >> 5;
  const int wm   = wave & 1;          // 2 wave rows
  const int wn   = wave >> 1;         // 4 wave cols
  const int h    = lane >> 4;
  const int l    = lane & 15;

  const int blockM = blockIdx.x * 64;
  const int nBlock = blockIdx.y * 256;
  const int tmB = blockM >> 4;        // first of 4 A tiles
  const int tnB = nBlock >> 4;        // first of 16 B tiles
  const int mBase = blockM + wm * 32;
  const int nBase = nBlock + wn * 64;

  const unsigned stageLo = (unsigned)(uintptr_t)&sh.stage[0][0];

  // Staging: 1280 16B tasks/chunk, 5 per thread.
  //   q=0  -> A region: tile = tid>>6 (4 tiles), sub = tid&63
  //   q>=1 -> B region: tile = (q-1)*4 + (tid>>6), sub = tid&63
  // Source deltas per q are the constant (q-1)*131072 B; LDS deltas (q-1)*4096 B.
  const char* gA = (const char*)(wsA + (size_t)(tmB + (tid >> 6)) * KC * 512) + (tid & 63) * 16;
  const char* gB = (const char*)(wsW + (size_t)(tnB + (tid >> 6)) * KC * 512) + (tid & 63) * 16;
  const unsigned ldsA = stageLo + (unsigned)tid * 16;
  const unsigned ldsB = stageLo + 4096u + (unsigned)tid * 16;

  // Stage current chunk into buffer `bufB` (byte offset) and advance pointers.
  auto stage = [&](unsigned bufB) {
    async_copy_b128(ldsA + bufB, gA);
    async_copy_b128(ldsB + bufB,          gB);
    async_copy_b128(ldsB + bufB + 4096u,  gB + 131072);
    async_copy_b128(ldsB + bufB + 8192u,  gB + 262144);
    async_copy_b128(ldsB + bufB + 12288u, gB + 393216);
    gA += 1024;                 // next K-chunk
    gB += 1024;
  };

  v8f c00 = {}, c01 = {}, c02 = {}, c03 = {};
  v8f c10 = {}, c11 = {}, c12 = {}, c13 = {};

  // LDS fragment offsets (elements): layout A[4 tiles] | B[16 tiles]
  const int aOff = wm * 1024 + lane * 16;          // tiles wm*2, wm*2+1
  const int bOff = 2048 + wn * 2048 + lane * 16;   // tiles wn*4 + j

  auto compute = [&](int buf) {
    const __bf16* sb = sh.stage[buf];
    v16bf a0 = *(const v16bf*)(sb + aOff);
    v16bf a1 = *(const v16bf*)(sb + aOff + 512);
    v16bf b0 = *(const v16bf*)(sb + bOff);
    v16bf b1 = *(const v16bf*)(sb + bOff + 512);
    v16bf b2 = *(const v16bf*)(sb + bOff + 1024);
    v16bf b3 = *(const v16bf*)(sb + bOff + 1536);
    c00 = __builtin_amdgcn_wmma_f32_16x16x32_bf16(false, a0, false, b0, (short)0, c00, false, false);
    c10 = __builtin_amdgcn_wmma_f32_16x16x32_bf16(false, a1, false, b0, (short)0, c10, false, false);
    c01 = __builtin_amdgcn_wmma_f32_16x16x32_bf16(false, a0, false, b1, (short)0, c01, false, false);
    c11 = __builtin_amdgcn_wmma_f32_16x16x32_bf16(false, a1, false, b1, (short)0, c11, false, false);
    c02 = __builtin_amdgcn_wmma_f32_16x16x32_bf16(false, a0, false, b2, (short)0, c02, false, false);
    c12 = __builtin_amdgcn_wmma_f32_16x16x32_bf16(false, a1, false, b2, (short)0, c12, false, false);
    c03 = __builtin_amdgcn_wmma_f32_16x16x32_bf16(false, a0, false, b3, (short)0, c03, false, false);
    c13 = __builtin_amdgcn_wmma_f32_16x16x32_bf16(false, a1, false, b3, (short)0, c13, false, false);
  };

  // Software pipeline: stage(0) -> [wait; barrier; stage(next); compute(cur)] x KC.
  // The final stage call prefetches 1 KB past the last chunk into the dead
  // buffer (still inside d_ws) -- branch-free loop body.
  stage(0u);
#pragma unroll 1
  for (int kc = 0; kc < KC; kc += 2) {
    wait_asynccnt0();
    __syncthreads();
    stage((unsigned)STAGE_BYTES);      // chunk kc+1 -> buffer 1
    compute(0);                        // chunk kc from buffer 0

    wait_asynccnt0();
    __syncthreads();
    stage(0u);                         // chunk kc+2 (or overshoot) -> buffer 0
    compute(1);                        // chunk kc+1 from buffer 1
  }
  wait_asynccnt0();                    // drain overshoot stage before LDS reuse
  __syncthreads();

  // ---------------- Epilogue ----------------
  v8f cr0[4] = {c00, c01, c02, c03};
  v8f cr1[4] = {c10, c11, c12, c13};

  // C/D layout: VGPR r -> row M = r + 8*h; lane l -> col N = l
#pragma unroll
  for (int j = 0; j < 4; ++j) {
    const int n   = nBase + j * 16 + l;
    const bool ok = (n < VOCAB);
    const float bv = ok ? bias[n] : 0.0f;
    const int colB = wn * 64 + j * 16 + l;       // block-local column 0..255
#pragma unroll
    for (int r = 0; r < 8; ++r) {
      const int rowB0 = wm * 32 + r + 8 * h;
      const int rowB1 = rowB0 + 16;
      const float v0 = cr0[j][r] + bv;
      const float v1 = cr1[j][r] + bv;
      sh.ep.vals[rowB0][colB] = ok ? v0 : -INFINITY;
      sh.ep.vals[rowB1][colB] = ok ? v1 : -INFINITY;
      if (ok) {
        const int mg = mBase + r + 8 * h;
        __builtin_nontemporal_store(v0, &logits[(size_t)mg * VOCAB + n]);
        __builtin_nontemporal_store(v1, &logits[(size_t)(mg + 16) * VOCAB + n]);
      }
    }
  }
  __syncthreads();

  // Per-row partial LSE over this block's 256 columns: 4 threads per row.
  {
    const int row = tid >> 2;
    const int seg = tid & 3;
    float m = -INFINITY, s = 0.0f;
    for (int i = 0; i < 64; ++i)
      lse_merge(m, s, sh.ep.vals[row][seg * 64 + i], 1.0f);
    sh.ep.pm[row][seg] = m;
    sh.ep.ps[row][seg] = s;
  }
  __syncthreads();

  if (tid < 64) {
    const int row = tid;
    float m = sh.ep.pm[row][0], s = sh.ep.ps[row][0];
    lse_merge(m, s, sh.ep.pm[row][1], sh.ep.ps[row][1]);
    lse_merge(m, s, sh.ep.pm[row][2], sh.ep.ps[row][2]);
    lse_merge(m, s, sh.ep.pm[row][3], sh.ep.ps[row][3]);
    float* p = partials + ((size_t)(blockM + row) * NBLK_N + blockIdx.y) * 2;
    p[0] = m;
    p[1] = s;
  }
}

// ---------------------------------------------------------------------------
// Merge NBLK_N partials per row + target logit -> rowLoss[row] = lse - tgt
// ---------------------------------------------------------------------------
__global__ __launch_bounds__(64)
void bigram_row_merge(const float* __restrict__ partials,
                      const float* __restrict__ logits,
                      const int* __restrict__ targets,
                      float* __restrict__ rowLoss) {
  __shared__ float sm[64];
  __shared__ float ss[64];
  const int row = blockIdx.x;
  const float* p = partials + (size_t)row * NBLK_N * 2;

  float m = -INFINITY, s = 0.0f;
  for (int j = threadIdx.x; j < NBLK_N; j += 64)
    lse_merge(m, s, p[2 * j], p[2 * j + 1]);
  sm[threadIdx.x] = m;
  ss[threadIdx.x] = s;
  __syncthreads();
  for (int step = 32; step > 0; step >>= 1) {
    if (threadIdx.x < step) {
      float mm = sm[threadIdx.x], sv = ss[threadIdx.x];
      lse_merge(mm, sv, sm[threadIdx.x + step], ss[threadIdx.x + step]);
      sm[threadIdx.x] = mm;
      ss[threadIdx.x] = sv;
    }
    __syncthreads();
  }
  if (threadIdx.x == 0) {
    float lse = sm[0] + __logf(ss[0]);
    float tgt = logits[(size_t)row * VOCAB + targets[row]];
    rowLoss[row] = lse - tgt;
  }
}

// ---------------------------------------------------------------------------
// Deterministic final mean reduction -> out[0]
// ---------------------------------------------------------------------------
__global__ __launch_bounds__(256)
void bigram_final_loss(const float* __restrict__ rowLoss, float* __restrict__ out) {
  __shared__ float sd[256];
  float s = 0.0f;
  for (int j = threadIdx.x; j < NTOK; j += 256) s += rowLoss[j];
  sd[threadIdx.x] = s;
  __syncthreads();
  for (int step = 128; step > 0; step >>= 1) {
    if (threadIdx.x < step) sd[threadIdx.x] += sd[threadIdx.x + step];
    __syncthreads();
  }
  if (threadIdx.x == 0) out[0] = sd[0] / (float)NTOK;
}

extern "C" void kernel_launch(void* const* d_in, const int* in_sizes, int n_in,
                              void* d_out, int out_size, void* d_ws, size_t ws_size,
                              hipStream_t stream) {
  const int*   idx     = (const int*)d_in[0];
  const int*   targets = (const int*)d_in[1];
  const float* emb     = (const float*)d_in[2];
  const float* W       = (const float*)d_in[3];
  const float* bias    = (const float*)d_in[4];

  float* logits = (float*)d_out;

  // Workspace layout
  const size_t wsA_bytes  = (size_t)NTOK * DM * sizeof(__bf16);            // 16.78 MB
  const size_t wsW_bytes  = (size_t)NPAD * DM * sizeof(__bf16);            // 103.3 MB
  const size_t part_bytes = (size_t)NTOK * NBLK_N * 2 * sizeof(float);     // 12.9 MB
  __bf16* wsA     = (__bf16*)d_ws;
  __bf16* wsW     = (__bf16*)((char*)d_ws + wsA_bytes);
  float* partials = (float*)((char*)d_ws + wsA_bytes + wsW_bytes);
  float* rowLoss  = (float*)((char*)d_ws + wsA_bytes + wsW_bytes + part_bytes);

  // One-time bf16 pre-swizzle (fragment-layout) conversions
  conv_x_bf16<<<(NTILES_M * KC * 32) / 256, 256, 0, stream>>>(idx, emb, wsA);
  conv_w_bf16<<<(NTILES_N * KC * 32) / 256, 256, 0, stream>>>(W, wsW);

  // GEMM + fused softmax partials. M-tile fast-varying -> W band L2-resident.
  dim3 grid(NTOK / 64, NBLK_N);
  bigram_logits_wmma<<<grid, 256, 0, stream>>>(wsA, wsW, bias, logits, partials);

  bigram_row_merge<<<NTOK, 64, 0, stream>>>(partials, logits, targets, rowLoss);

  bigram_final_loss<<<1, 256, 0, stream>>>(rowLoss, logits + (size_t)NTOK * VOCAB);
}
